// Bayesian_NN_Moe_54065048322710
// MI455X (gfx1250) — matrix operand order repaired and checked by reading, not compiled, not run on hardware
//
#include <hip/hip_runtime.h>
#include <hip/hip_bf16.h>

// Problem constants
#define BB_   8192   // batch (tokens)
#define DIN_  512
#define F_    1024
#define H_    2048
#define C_    1024
#define E_    32
#define CAP_  640
#define PI8_  0.39269908169872414f

#define LDA_  40     // padded LDS stride (f16 elems) for A rows and packed-B cols

typedef __attribute__((ext_vector_type(16))) _Float16 v16h;
typedef __attribute__((ext_vector_type(8)))  float    v8f;

// ---------------------------------------------------------------------------
// WMMA operand loaders (wave32 layouts per CDNA5 ISA 7.12.2)
// A 16x32 f16: lane L<16 -> row M=L, K 0..15; lane L>=16 -> row M=L-16, K 16..31 (contiguous k, same as B)
// B packed in LDS column-major [n][k] so each lane reads 16 contiguous f16.
// C/D 16x16 f32: v8f element r -> M = (lane>>4)*8 + r, N = lane&15
// ---------------------------------------------------------------------------
__device__ __forceinline__ v16h lds_load_A(const _Float16* base) {
    int lane = threadIdx.x & 31;
    int row  = lane & 15;
    int kb   = (lane >> 4) << 4;          // 0 or 16
    const _Float16* p = base + row * LDA_;
    v16h a;
#pragma unroll
    for (int i = 0; i < 8; ++i) a[i] = p[kb + i];
#pragma unroll
    for (int i = 0; i < 8; ++i) a[8 + i] = p[kb + 8 + i];
    return a;
}

__device__ __forceinline__ v16h lds_load_Bpack(const _Float16* base) {
    int lane = threadIdx.x & 31;
    const _Float16* p = base + (lane & 15) * LDA_ + ((lane >> 4) << 4);
    v16h b;
#pragma unroll
    for (int i = 0; i < 16; ++i) b[i] = p[i];   // 32B contiguous -> 2x ds_load_b128
    return b;
}

#define WMMA_F16(A, B, Cacc) \
    __builtin_amdgcn_wmma_f32_16x16x32_f16(false, (A), false, (B), (short)0, (Cacc), false, false)

// CDNA5 async global->LDS copy (16B per lane) + fence
__device__ __forceinline__ void async_copy_b128(void* lds_dst, const void* gsrc) {
    unsigned off = (unsigned)(uintptr_t)lds_dst;     // LDS aperture: low 32 bits = LDS offset
    asm volatile("global_load_async_to_lds_b128 %0, %1, off"
                 :: "v"(off), "v"(gsrc) : "memory");
}
__device__ __forceinline__ void async_wait0() {
    asm volatile("s_wait_asynccnt 0" ::: "memory");
}

// Stage a 32(K)x128(N) B tile from f32 global (row-major [k][n]) into packed
// column-major LDS [n][k] with f32->f16 conversion. 256 threads.
__device__ __forceinline__ void stage_B_pack(const float* __restrict__ src,
                                             size_t ldg, _Float16* Bp, int tid) {
    int k   = (tid >> 4) << 1;           // even k: 0..30
    int nst = (tid & 15) << 3;           // 0..120
    const float* s0 = src + (size_t)k * ldg + nst;
    const float* s1 = s0 + ldg;
#pragma unroll
    for (int j = 0; j < 8; ++j) {
        union { _Float16 h[2]; unsigned u; } p;
        p.h[0] = (_Float16)s0[j];
        p.h[1] = (_Float16)s1[j];
        *(unsigned*)&Bp[(nst + j) * LDA_ + k] = p.u;   // 4B aligned
    }
}

// ---------------------------------------------------------------------------
// Init: zero output + dispatch tables, precompute exp(W_logvar)
// ---------------------------------------------------------------------------
__global__ __launch_bounds__(256) void init_kernel(
    float* __restrict__ out, int outN,
    int* __restrict__ cnt, int* __restrict__ tok, float* __restrict__ vals, int slotN,
    const float* __restrict__ W_logvar, float* __restrict__ expW, int wN) {
    int i = blockIdx.x * blockDim.x + threadIdx.x;
    if (i < outN)  out[i] = 0.0f;
    if (i < slotN) { tok[i] = 0; vals[i] = 0.0f; }
    if (i < E_)    cnt[i] = 0;
    if (i < wN)    expW[i] = __expf(W_logvar[i]);
}

// ---------------------------------------------------------------------------
// Backbone GEMM: h = x[8192,512] @ Wb[512,1024] + bb  -> h_f32 and h_f16
// Block tile 128x128, BK=32, 8 waves in 4x2, each wave 32x64 via 2x4 WMMA
// ---------------------------------------------------------------------------
__global__ __launch_bounds__(256) void backbone_gemm(
    const float* __restrict__ x, const float* __restrict__ Wb,
    const float* __restrict__ bb,
    float* __restrict__ hout, _Float16* __restrict__ houth) {
    const int K = DIN_, N = F_;
    __shared__ __align__(16) _Float16 As[128 * LDA_];
    __shared__ __align__(16) _Float16 Bp[128 * LDA_];
    const int tid  = threadIdx.x;
    const int wave = tid >> 5;
    const int wm = wave >> 1, wn = wave & 1;
    const int bm = blockIdx.y * 128, bn = blockIdx.x * 128;
    v8f c[2][4] = {};

    for (int k0 = 0; k0 < K; k0 += 32) {
        {   // stage A (128x32), f32 -> f16
            int row = tid >> 1;
            int kst = (tid & 1) << 4;
            const float* src = x + (size_t)(bm + row) * K + k0 + kst;
            _Float16* dst = As + row * LDA_ + kst;
#pragma unroll
            for (int j = 0; j < 16; ++j) dst[j] = (_Float16)src[j];
        }
        stage_B_pack(Wb + (size_t)k0 * N + bn, N, Bp, tid);
        __syncthreads();
        v16h a0 = lds_load_A(As + (wm * 32 + 0)  * LDA_);
        v16h a1 = lds_load_A(As + (wm * 32 + 16) * LDA_);
        v16h b0 = lds_load_Bpack(Bp + (wn * 64 + 0)  * LDA_);
        v16h b1 = lds_load_Bpack(Bp + (wn * 64 + 16) * LDA_);
        v16h b2 = lds_load_Bpack(Bp + (wn * 64 + 32) * LDA_);
        v16h b3 = lds_load_Bpack(Bp + (wn * 64 + 48) * LDA_);
        c[0][0] = WMMA_F16(a0, b0, c[0][0]);
        c[0][1] = WMMA_F16(a0, b1, c[0][1]);
        c[0][2] = WMMA_F16(a0, b2, c[0][2]);
        c[0][3] = WMMA_F16(a0, b3, c[0][3]);
        c[1][0] = WMMA_F16(a1, b0, c[1][0]);
        c[1][1] = WMMA_F16(a1, b1, c[1][1]);
        c[1][2] = WMMA_F16(a1, b2, c[1][2]);
        c[1][3] = WMMA_F16(a1, b3, c[1][3]);
        __syncthreads();
    }

    int lane = tid & 31;
    int nin  = lane & 15;
    int mb   = (lane >> 4) << 3;
#pragma unroll
    for (int i = 0; i < 2; ++i)
#pragma unroll
        for (int j = 0; j < 4; ++j) {
            int col = bn + wn * 64 + j * 16 + nin;
            float bias = bb[col];
#pragma unroll
            for (int r = 0; r < 8; ++r) {
                int row = bm + wm * 32 + i * 16 + mb + r;
                float v = c[i][j][r] + bias;
                hout[(size_t)row * N + col]  = v;
                houth[(size_t)row * N + col] = (_Float16)v;
            }
        }
}

// ---------------------------------------------------------------------------
// Router: per-token mu/var -> logistic-normal shrink -> softmax top-2 ->
// renormalize -> capacity dispatch via atomic slot counters. 128 thr/token.
// ---------------------------------------------------------------------------
__global__ __launch_bounds__(128) void router_kernel(
    const float* __restrict__ h, const float* __restrict__ W_mu,
    const float* __restrict__ expW, const float* __restrict__ b_mu,
    const float* __restrict__ b_logvar,
    int* __restrict__ cnt, int* __restrict__ tok, float* __restrict__ vals) {
    __shared__ float hrow[F_];
    __shared__ float mup[128], varp[128];
    __shared__ float tilde[E_];
    const int t = blockIdx.x;
    const int tid = threadIdx.x;

    const float* hp = h + (size_t)t * F_;
    for (int i = tid; i < F_; i += 128) hrow[i] = hp[i];
    __syncthreads();

    int e = tid & 31, s = tid >> 5;       // 32 experts x 4 K-slices of 256
    float m = 0.f, v = 0.f;
    const float* wm = W_mu + (size_t)e * F_;
    const float* wv = expW + (size_t)e * F_;
    for (int k = s * 256; k < s * 256 + 256; ++k) {
        float hk = hrow[k];
        m = fmaf(hk, wm[k], m);
        v = fmaf(hk * hk, wv[k], v);
    }
    mup[tid] = m; varp[tid] = v;
    __syncthreads();

    if (tid < E_) {
        float mm = mup[tid] + mup[tid + 32] + mup[tid + 64] + mup[tid + 96] + b_mu[tid];
        float vv = varp[tid] + varp[tid + 32] + varp[tid + 64] + varp[tid + 96]
                 + __expf(b_logvar[tid]);
        vv = fmaxf(vv, 1e-12f);
        tilde[tid] = mm * rsqrtf(1.0f + PI8_ * vv);
    }
    __syncthreads();

    if (tid == 0) {
        float best = -1e30f; int bi = 0;
        for (int i = 0; i < E_; ++i) if (tilde[i] > best) { best = tilde[i]; bi = i; }
        float best2 = -1e30f; int bi2 = (bi == 0) ? 1 : 0;
        for (int i = 0; i < E_; ++i)
            if (i != bi && tilde[i] > best2) { best2 = tilde[i]; bi2 = i; }
        float sum = 0.f;
        for (int i = 0; i < E_; ++i) sum += __expf(tilde[i] - best);
        float p1 = 1.0f / sum;
        float p2 = __expf(best2 - best) / sum;
        float den = fmaxf(p1 + p2, 1e-12f);
        float w1 = p1 / den, w2 = p2 / den;
        int pos1 = atomicAdd(&cnt[bi], 1);
        if (pos1 < CAP_) { tok[bi * CAP_ + pos1] = t; vals[bi * CAP_ + pos1] = w1; }
        int pos2 = atomicAdd(&cnt[bi2], 1);
        if (pos2 < CAP_) { tok[bi2 * CAP_ + pos2] = t; vals[bi2 * CAP_ + pos2] = w2; }
    }
}

// ---------------------------------------------------------------------------
// Expert layer 1: z[e] = relu(gather(h)[tok] @ W1[e] + b1[e])  [640,2048]
// A gathered via tok indices from h_f16 using async global->LDS B128 copies.
// ---------------------------------------------------------------------------
__global__ __launch_bounds__(256) void expert_gemm1(
    const _Float16* __restrict__ hh, const float* __restrict__ W1,
    const float* __restrict__ b1, const int* __restrict__ tok,
    _Float16* __restrict__ z) {
    const int K = F_, N = H_;
    __shared__ __align__(16) _Float16 As[128 * LDA_];
    __shared__ __align__(16) _Float16 Bp[128 * LDA_];
    const int tid  = threadIdx.x;
    const int wave = tid >> 5;
    const int wm = wave >> 1, wn = wave & 1;
    const int e  = blockIdx.z;
    const int bm = blockIdx.y * 128, bn = blockIdx.x * 128;
    const float* W1e = W1 + (size_t)e * K * N;
    const int arow = tid >> 1;
    const int ahalf = (tid & 1) << 4;                 // f16 element offset 0/16
    const int tix = tok[e * CAP_ + bm + arow];
    const _Float16* arow_src = hh + (size_t)tix * K + ahalf;
    _Float16* arow_dst = As + arow * LDA_ + ahalf;
    v8f c[2][4] = {};

    for (int k0 = 0; k0 < K; k0 += 32) {
        // async gather-stage A (128x32 f16): 2x16B per thread
        async_copy_b128(arow_dst,     arow_src + k0);
        async_copy_b128(arow_dst + 8, arow_src + k0 + 8);
        {   // stage B tile of W1[e], f32 -> f16, packed column-major
            const float* src = W1e + (size_t)k0 * N + bn;
            if (k0 + 32 < K) __builtin_prefetch(src + (size_t)32 * N, 0, 1);
            stage_B_pack(src, N, Bp, tid);
        }
        async_wait0();
        __syncthreads();
        v16h a0 = lds_load_A(As + (wm * 32 + 0)  * LDA_);
        v16h a1 = lds_load_A(As + (wm * 32 + 16) * LDA_);
        v16h b0 = lds_load_Bpack(Bp + (wn * 64 + 0)  * LDA_);
        v16h b1 = lds_load_Bpack(Bp + (wn * 64 + 16) * LDA_);
        v16h b2 = lds_load_Bpack(Bp + (wn * 64 + 32) * LDA_);
        v16h b3 = lds_load_Bpack(Bp + (wn * 64 + 48) * LDA_);
        c[0][0] = WMMA_F16(a0, b0, c[0][0]);
        c[0][1] = WMMA_F16(a0, b1, c[0][1]);
        c[0][2] = WMMA_F16(a0, b2, c[0][2]);
        c[0][3] = WMMA_F16(a0, b3, c[0][3]);
        c[1][0] = WMMA_F16(a1, b0, c[1][0]);
        c[1][1] = WMMA_F16(a1, b1, c[1][1]);
        c[1][2] = WMMA_F16(a1, b2, c[1][2]);
        c[1][3] = WMMA_F16(a1, b3, c[1][3]);
        __syncthreads();
    }

    int lane = tid & 31;
    int nin  = lane & 15;
    int mb   = (lane >> 4) << 3;
    _Float16* ze = z + (size_t)e * CAP_ * N;
#pragma unroll
    for (int i = 0; i < 2; ++i)
#pragma unroll
        for (int j = 0; j < 4; ++j) {
            int col = bn + wn * 64 + j * 16 + nin;
            float bias = b1[e * N + col];
#pragma unroll
            for (int r = 0; r < 8; ++r) {
                int row = bm + wm * 32 + i * 16 + mb + r;
                float v = fmaxf(c[i][j][r] + bias, 0.0f);
                ze[(size_t)row * N + col] = (_Float16)v;
            }
        }
}

// ---------------------------------------------------------------------------
// Expert layer 2 + combine: y = z[e] @ W2[e] + b2[e]; out[tok] += vals * y
// ---------------------------------------------------------------------------
__global__ __launch_bounds__(256) void expert_gemm2(
    const _Float16* __restrict__ z, const float* __restrict__ W2,
    const float* __restrict__ b2, const int* __restrict__ tok,
    const float* __restrict__ vals, float* __restrict__ out) {
    const int K = H_, N = C_;
    __shared__ __align__(16) _Float16 As[128 * LDA_];
    __shared__ __align__(16) _Float16 Bp[128 * LDA_];
    const int tid  = threadIdx.x;
    const int wave = tid >> 5;
    const int wm = wave >> 1, wn = wave & 1;
    const int e  = blockIdx.z;
    const int bm = blockIdx.y * 128, bn = blockIdx.x * 128;
    const float* W2e = W2 + (size_t)e * K * N;
    const int arow = tid >> 1;
    const int ahalf = (tid & 1) << 4;
    const _Float16* arow_src = z + ((size_t)e * CAP_ + bm + arow) * K + ahalf;
    _Float16* arow_dst = As + arow * LDA_ + ahalf;
    v8f c[2][4] = {};

    for (int k0 = 0; k0 < K; k0 += 32) {
        // async stage A from z (f16 direct): 2x16B per thread
        async_copy_b128(arow_dst,     arow_src + k0);
        async_copy_b128(arow_dst + 8, arow_src + k0 + 8);
        {   // stage B tile of W2[e], f32 -> f16, packed column-major
            const float* src = W2e + (size_t)k0 * N + bn;
            if (k0 + 32 < K) __builtin_prefetch(src + (size_t)32 * N, 0, 1);
            stage_B_pack(src, N, Bp, tid);
        }
        async_wait0();
        __syncthreads();
        v16h a0 = lds_load_A(As + (wm * 32 + 0)  * LDA_);
        v16h a1 = lds_load_A(As + (wm * 32 + 16) * LDA_);
        v16h b0 = lds_load_Bpack(Bp + (wn * 64 + 0)  * LDA_);
        v16h b1 = lds_load_Bpack(Bp + (wn * 64 + 16) * LDA_);
        v16h b2 = lds_load_Bpack(Bp + (wn * 64 + 32) * LDA_);
        v16h b3 = lds_load_Bpack(Bp + (wn * 64 + 48) * LDA_);
        c[0][0] = WMMA_F16(a0, b0, c[0][0]);
        c[0][1] = WMMA_F16(a0, b1, c[0][1]);
        c[0][2] = WMMA_F16(a0, b2, c[0][2]);
        c[0][3] = WMMA_F16(a0, b3, c[0][3]);
        c[1][0] = WMMA_F16(a1, b0, c[1][0]);
        c[1][1] = WMMA_F16(a1, b1, c[1][1]);
        c[1][2] = WMMA_F16(a1, b2, c[1][2]);
        c[1][3] = WMMA_F16(a1, b3, c[1][3]);
        __syncthreads();
    }

    int lane = tid & 31;
    int nin  = lane & 15;
    int mb   = (lane >> 4) << 3;
#pragma unroll
    for (int i = 0; i < 2; ++i)
#pragma unroll
        for (int j = 0; j < 4; ++j) {
            int col = bn + wn * 64 + j * 16 + nin;
            float bias = b2[e * N + col];
#pragma unroll
            for (int r = 0; r < 8; ++r) {
                int row = bm + wm * 32 + i * 16 + mb + r;
                float w = vals[e * CAP_ + row];
                if (w != 0.0f) {
                    int tix = tok[e * CAP_ + row];
                    atomicAdd(&out[(size_t)tix * N + col], w * (c[i][j][r] + bias));
                }
            }
        }
}

// ---------------------------------------------------------------------------
extern "C" void kernel_launch(void* const* d_in, const int* in_sizes, int n_in,
                              void* d_out, int out_size, void* d_ws, size_t ws_size,
                              hipStream_t stream) {
    const float* x        = (const float*)d_in[0];
    const float* Wb       = (const float*)d_in[1];
    const float* bb       = (const float*)d_in[2];
    const float* W_mu     = (const float*)d_in[3];
    const float* W_logvar = (const float*)d_in[4];
    const float* b_mu     = (const float*)d_in[5];
    const float* b_logvar = (const float*)d_in[6];
    const float* W1       = (const float*)d_in[7];
    const float* b1       = (const float*)d_in[8];
    const float* W2       = (const float*)d_in[9];
    const float* b2       = (const float*)d_in[10];
    float* out = (float*)d_out;

    // Workspace carve-up (~134.5 MB)
    char* w = (char*)d_ws;
    float*    h_f32 = (float*)w;     w += (size_t)BB_ * F_ * 4;
    _Float16* h_f16 = (_Float16*)w;  w += (size_t)BB_ * F_ * 2;
    _Float16* z_f16 = (_Float16*)w;  w += (size_t)E_ * CAP_ * H_ * 2;
    float*    expW  = (float*)w;     w += (size_t)E_ * F_ * 4;
    float*    vals  = (float*)w;     w += (size_t)E_ * CAP_ * 4;
    int*      tok   = (int*)w;       w += (size_t)E_ * CAP_ * 4;
    int*      cnt   = (int*)w;       w += (size_t)E_ * 4;

    const int outN  = BB_ * C_;
    const int slotN = E_ * CAP_;
    const int wN    = E_ * F_;

    init_kernel<<<(outN + 255) / 256, 256, 0, stream>>>(
        out, outN, cnt, tok, vals, slotN, W_logvar, expW, wN);

    backbone_gemm<<<dim3(F_ / 128, BB_ / 128), 256, 0, stream>>>(
        x, Wb, bb, h_f32, h_f16);

    router_kernel<<<BB_, 128, 0, stream>>>(
        h_f32, W_mu, expW, b_mu, b_logvar, cnt, tok, vals);

    expert_gemm1<<<dim3(H_ / 128, CAP_ / 128, E_), 256, 0, stream>>>(
        h_f16, W1, b1, tok, z_f16);

    expert_gemm2<<<dim3(C_ / 128, CAP_ / 128, E_), 256, 0, stream>>>(
        z_f16, W2, b2, tok, vals, out);
}
